// Attention_67491116089608
// MI455X (gfx1250) — matrix operand order repaired and checked by reading, not compiled, not run on hardware
//
#include <hip/hip_runtime.h>
#include <hip/hip_bf16.h>

// ---------------------------------------------------------------------------
// MI455X (gfx1250) attention: bf16 WMMA GEMMs + f32 online softmax.
//
// Pipeline (all in d_ws, bf16 where possible):
//   1. x  (f32)          -> xbf  [B*S][D] bf16
//   2. Wq/Wk/Wv (f32)    -> Wt*  [D][D]   bf16, transposed to [out][in]
//   3. Q/K/V = xbf @ W + b       (WMMA bf16, f32 accum, bf16 out)
//   4. Vbf -> Vt [D][S] bf16 (B operand for P@V must be K-contiguous per col)
//   5. scores pass: S = Q @ K^T (WMMA), raw bf16 scores to Sbuf,
//      online row max m / row sum-of-exp l in f32 (wave shfl reductions)
//   6. normalize: P = exp(S - m) / l, bf16 in place
//   7. out = P @ V (WMMA bf16, f32 accum, f32 out)
//
// GEMM/scores inner loops: K_STEP=64, double-buffered LDS, staged with
// GLOBAL_LOAD_ASYNC_TO_LDS_B128 (ASYNCcnt) when the builtin exists, else
// global_load_b128 + ds_store_b128.
// ---------------------------------------------------------------------------

typedef __bf16 v4bf  __attribute__((ext_vector_type(4)));
typedef __bf16 v8bf  __attribute__((ext_vector_type(8)));
typedef __bf16 v16bf __attribute__((ext_vector_type(16)));
typedef float  v8f   __attribute__((ext_vector_type(8)));
typedef int    v4i   __attribute__((ext_vector_type(4)));

#define KSTEP 64
#define LDT   72  // 64 bf16 payload + 8 pad -> 144B rows: 16B aligned, conflict-free

// ---- Async Global->LDS staging (CDNA5) ------------------------------------
#if __has_builtin(__builtin_amdgcn_global_load_async_to_lds_b128)
#define HAVE_ASYNC_LDS 1
#else
#define HAVE_ASYNC_LDS 0
#endif

// Typed address-space pointers (the builtin takes v4i AS1/AS3 pointers).
typedef __attribute__((address_space(1))) v4i* gv4i_ptr;
typedef __attribute__((address_space(3))) v4i* lv4i_ptr;

__device__ inline void async_copy16(__bf16* lds, const __bf16* g) {
#if HAVE_ASYNC_LDS
  // Flat->AS casts via integers: AS1 value == flat address; AS3 value ==
  // flat LDS address low 32 bits (ISA 10.2 aperture mapping).
  __builtin_amdgcn_global_load_async_to_lds_b128(
      (gv4i_ptr)(uintptr_t)g,
      (lv4i_ptr)(uint32_t)(uintptr_t)lds,
      0, 0);
#else
  *(v8bf*)lds = *(const v8bf*)g;  // global_load_b128 + ds_store_b128
#endif
}

#if HAVE_ASYNC_LDS
#if __has_builtin(__builtin_amdgcn_s_wait_asynccnt)
#define WAIT_ASYNC() __builtin_amdgcn_s_wait_asynccnt(0)
#else
#define WAIT_ASYNC() asm volatile("s_wait_asynccnt 0x0" ::: "memory")
#endif
#else
#define WAIT_ASYNC() ((void)0)
#endif

// ---- WMMA wrapper ---------------------------------------------------------
__device__ inline v8f wmma_bf16(v16bf a, v16bf b, v8f c) {
  // D = A(16x32 bf16) * B(32x16 bf16) + C(16x16 f32)
  return __builtin_amdgcn_wmma_f32_16x16x32_bf16(
      /*neg_a=*/false, a, /*neg_b=*/false, b,
      /*c_mod=*/(short)0, c, /*reuse_a=*/false, /*reuse_b=*/false);
}

// ---- Fragment loaders (layouts per CDNA5 ISA §7.12.2) ----------
// A fragment, 16x32 bf16 from row-major LDS tile (stride in elements).
// Lane holds row M = lane&15; elems 0..7 = K[8h..8h+7], elems 8..15 = K[16+8h..]
__device__ inline v16bf load_frag_a(const __bf16* base, int stride, int lane) {
  int m = lane & 15, h = lane >> 4;
  const __bf16* row = base + (size_t)m * stride;
  v8bf lo = *(const v8bf*)(row + 8 * h);
  v8bf hi = *(const v8bf*)(row + 16 + 8 * h);
  return __builtin_shufflevector(lo, hi, 0,1,2,3,4,5,6,7,8,9,10,11,12,13,14,15);
}

// B fragment, 32x16 bf16, sourced from K-contiguous [N][K] storage in LDS.
// Lane holds col N = lane&15; elems 0..7 = K[8h..8h+7], elems 8..15 = K[16+8h..]
__device__ inline v16bf load_frag_b(const __bf16* base, int stride, int lane) {
  int n = lane & 15, h = lane >> 4;
  const __bf16* row = base + (size_t)n * stride + 8 * h;
  v8bf lo = *(const v8bf*)(row);
  v8bf hi = *(const v8bf*)(row + 16);
  return __builtin_shufflevector(lo, hi, 0,1,2,3,4,5,6,7,8,9,10,11,12,13,14,15);
}

// Cooperative 256-thread staging of a ROWSx64 bf16 tile global->LDS.
template<int ROWS>
__device__ inline void stage_tile(__bf16* lds, const __bf16* g, int ldg) {
  const int chunks = ROWS * (KSTEP / 8);  // 8 x v8bf per 64-wide row
  for (int c = threadIdx.x; c < chunks; c += 256) {
    int r = c >> 3, o = (c & 7) * 8;
    async_copy16(lds + r * LDT + o, g + (size_t)r * ldg + o);
  }
}

// ---- Kernel: f32 -> bf16 elementwise (vectorized x4) ----------------------
__global__ __launch_bounds__(256)
void cvt_f32_to_bf16_vec(const float4* __restrict__ in, v4bf* __restrict__ out,
                         long n4) {
  long i  = (long)blockIdx.x * blockDim.x + threadIdx.x;
  long st = (long)gridDim.x * blockDim.x;
  for (; i < n4; i += st) {
    float4 f = in[i];
    v4bf o;
    o[0] = (__bf16)f.x; o[1] = (__bf16)f.y; o[2] = (__bf16)f.z; o[3] = (__bf16)f.w;
    out[i] = o;
  }
}

// ---- Kernel: W [in][out] f32 -> Wt [out][in] bf16 (LDS tile transpose) ----
__global__ __launch_bounds__(256)
void transpose_w_f32_to_bf16(const float* __restrict__ W, __bf16* __restrict__ Wt,
                             int Dd) {
  __shared__ float tile[32][33];
  int e0 = blockIdx.x * 32;  // out dim
  int d0 = blockIdx.y * 32;  // in dim
  int tx = threadIdx.x & 31, ty = threadIdx.x >> 5;
#pragma unroll
  for (int i = 0; i < 32; i += 8)
    tile[ty + i][tx] = W[(size_t)(d0 + ty + i) * Dd + e0 + tx];
  __syncthreads();
#pragma unroll
  for (int i = 0; i < 32; i += 8)
    Wt[(size_t)(e0 + ty + i) * Dd + d0 + tx] = (__bf16)tile[tx][ty + i];
}

// ---- Kernel: batched bf16 transpose [z][R][C] -> [z][C][R] ----------------
__global__ __launch_bounds__(256)
void transpose_bf16_batched(const __bf16* __restrict__ in, __bf16* __restrict__ out,
                            int R, int C) {
  __shared__ __bf16 tile[32][33];
  size_t zb = (size_t)blockIdx.z * R * C;
  int c0 = blockIdx.x * 32, r0 = blockIdx.y * 32;
  int tx = threadIdx.x & 31, ty = threadIdx.x >> 5;
#pragma unroll
  for (int i = 0; i < 32; i += 8)
    tile[ty + i][tx] = in[zb + (size_t)(r0 + ty + i) * C + c0 + tx];
  __syncthreads();
#pragma unroll
  for (int i = 0; i < 32; i += 8)
    out[zb + (size_t)(c0 + ty + i) * R + r0 + tx] = tile[tx][ty + i];
}

// ---- Kernel: bf16 WMMA GEMM  C[M][N] = A[M][K] * Bt[N][K]^T (+bias) -------
// Block = 256 threads (8 waves), tile 128x64, K_STEP 64, double-buffered LDS.
// Wave w -> rows 16w..16w+15, 4 column sub-tiles; 8 WMMAs per barrier.
__global__ __launch_bounds__(256)
void gemm_bf16_wmma(const __bf16* __restrict__ A, int lda, long strideA,
                    const __bf16* __restrict__ Bt, int ldb, long strideB,
                    const float* __restrict__ bias,
                    __bf16* __restrict__ outB, float* __restrict__ outF,
                    int ldc, long strideC, int K) {
  __shared__ __attribute__((aligned(16))) __bf16 ldsA[2][128 * LDT];
  __shared__ __attribute__((aligned(16))) __bf16 ldsB[2][64 * LDT];
  const int z = blockIdx.z;
  A  += (size_t)z * strideA;
  Bt += (size_t)z * strideB;
  const size_t cbase = (size_t)z * strideC;
  const int m0 = blockIdx.x * 128, n0 = blockIdx.y * 64;
  const int lane = threadIdx.x & 31, w = threadIdx.x >> 5;
  const __bf16* Ab = A + (size_t)m0 * lda;
  const __bf16* Bb = Bt + (size_t)n0 * ldb;

  v8f acc[4] = {};  // 4 x (16x16 f32) accumulators per wave

  const int nk = K / KSTEP;
  stage_tile<128>(ldsA[0], Ab, lda);
  stage_tile<64>(ldsB[0], Bb, ldb);
  WAIT_ASYNC();
  __syncthreads();

  for (int kt = 0; kt < nk; ++kt) {
    const int cur = kt & 1, nxt = cur ^ 1;
    if (kt + 1 < nk) {  // stage next slab while computing current
      stage_tile<128>(ldsA[nxt], Ab + (size_t)(kt + 1) * KSTEP, lda);
      stage_tile<64>(ldsB[nxt], Bb + (size_t)(kt + 1) * KSTEP, ldb);
    }
    if (kt + 2 < nk)  // cache hint two slabs ahead (global_prefetch_b8)
      __builtin_prefetch(Ab + (size_t)(threadIdx.x >> 1) * lda + (size_t)(kt + 2) * KSTEP, 0, 1);
#pragma unroll
    for (int kk = 0; kk < 2; ++kk) {
      v16bf a = load_frag_a(ldsA[cur] + 16 * w * LDT + kk * 32, LDT, lane);
#pragma unroll
      for (int n = 0; n < 4; ++n) {
        v16bf b = load_frag_b(ldsB[cur] + 16 * n * LDT + kk * 32, LDT, lane);
        acc[n] = wmma_bf16(a, b, acc[n]);
      }
    }
    WAIT_ASYNC();      // next-slab async writes landed
    __syncthreads();   // all waves done reading cur / writing nxt
  }

  // Epilogue: C layout — VGPR r holds row M = r + 8*(lane>>4), col N = lane&15.
  const int h = lane >> 4;
#pragma unroll
  for (int n = 0; n < 4; ++n) {
    int cn = n0 + 16 * n + (lane & 15);
    float bb = bias ? bias[cn] : 0.0f;
#pragma unroll
    for (int r = 0; r < 8; ++r) {
      int mr = m0 + 16 * w + r + 8 * h;
      float val = acc[n][r] + bb;
      size_t off = cbase + (size_t)mr * ldc + cn;
      if (outB) outB[off] = (__bf16)val;
      else      outF[off] = val;
    }
  }
}

// ---- Kernel: scores S = Q @ K^T + online softmax stats --------------------
// Grid: (S/128, 1, B). Wave w owns q-rows 16w..16w+15. Streams 64-key tiles;
// inner d-loop in 64-chunks, double-buffered. K is [key][d] row-major, which
// is exactly the K-contiguous B-operand layout.
__global__ __launch_bounds__(256)
void attn_scores_bf16(const __bf16* __restrict__ Q, const __bf16* __restrict__ Kmat,
                      __bf16* __restrict__ Sbuf, float* __restrict__ Mrow,
                      float* __restrict__ Lrow, int Sq, int Dd) {
  __shared__ __attribute__((aligned(16))) __bf16 ldsQ[2][128 * LDT];
  __shared__ __attribute__((aligned(16))) __bf16 ldsK[2][64 * LDT];
  const int bz = blockIdx.z;
  const int q0 = blockIdx.x * 128;
  const __bf16* Qb = Q    + ((size_t)bz * Sq + q0) * Dd;
  const __bf16* Kb = Kmat + (size_t)bz * Sq * Dd;
  const int lane = threadIdx.x & 31, w = threadIdx.x >> 5, h = lane >> 4;

  float m[8], l[8];
#pragma unroll
  for (int r = 0; r < 8; ++r) { m[r] = -__builtin_inff(); l[r] = 0.0f; }

  const int nd = Dd / KSTEP;
  for (int kt = 0; kt < Sq; kt += 64) {
    v8f acc[4] = {};
    const __bf16* Kt = Kb + (size_t)kt * Dd;

    stage_tile<128>(ldsQ[0], Qb, Dd);
    stage_tile<64>(ldsK[0], Kt, Dd);
    WAIT_ASYNC();
    __syncthreads();

    for (int dt = 0; dt < nd; ++dt) {
      const int cur = dt & 1, nxt = cur ^ 1;
      if (dt + 1 < nd) {
        stage_tile<128>(ldsQ[nxt], Qb + (size_t)(dt + 1) * KSTEP, Dd);
        stage_tile<64>(ldsK[nxt], Kt + (size_t)(dt + 1) * KSTEP, Dd);
      }
#pragma unroll
      for (int kk = 0; kk < 2; ++kk) {
        v16bf a = load_frag_a(ldsQ[cur] + 16 * w * LDT + kk * 32, LDT, lane);
#pragma unroll
        for (int n = 0; n < 4; ++n) {
          v16bf b = load_frag_b(ldsK[cur] + 16 * n * LDT + kk * 32, LDT, lane);
          acc[n] = wmma_bf16(a, b, acc[n]);
        }
      }
      WAIT_ASYNC();
      __syncthreads();
    }

    // Online softmax update + raw score store.
    // Row M = r + 8h lives in VGPR r across one 16-lane half; xor-shuffles
    // with masks 1..8 reduce within that half (wave32).
#pragma unroll
    for (int r = 0; r < 8; ++r) {
      float mx = fmaxf(fmaxf(acc[0][r], acc[1][r]), fmaxf(acc[2][r], acc[3][r]));
#pragma unroll
      for (int s = 1; s < 16; s <<= 1) mx = fmaxf(mx, __shfl_xor(mx, s, 32));
      float nm = fmaxf(m[r], mx);
      float se = __expf(acc[0][r] - nm) + __expf(acc[1][r] - nm) +
                 __expf(acc[2][r] - nm) + __expf(acc[3][r] - nm);
#pragma unroll
      for (int s = 1; s < 16; s <<= 1) se += __shfl_xor(se, s, 32);
      l[r] = l[r] * __expf(m[r] - nm) + se;
      m[r] = nm;

      int qrow = q0 + 16 * w + r + 8 * h;
      size_t base = ((size_t)bz * Sq + qrow) * Sq + kt + (lane & 15);
#pragma unroll
      for (int n = 0; n < 4; ++n) Sbuf[base + 16 * n] = (__bf16)acc[n][r];
    }
  }

  if ((lane & 15) == 0) {
#pragma unroll
    for (int r = 0; r < 8; ++r) {
      int qrow = q0 + 16 * w + r + 8 * h;
      Mrow[(size_t)bz * Sq + qrow] = m[r];
      Lrow[(size_t)bz * Sq + qrow] = l[r];
    }
  }
}

// ---- Kernel: P = exp(S - m) / l, bf16 in place (vectorized x8) ------------
__global__ __launch_bounds__(256)
void softmax_normalize(__bf16* __restrict__ Sbuf, const float* __restrict__ Mrow,
                       const float* __restrict__ Lrow, int Sq, long total) {
  long i  = ((long)blockIdx.x * blockDim.x + threadIdx.x) * 8;
  long st = (long)gridDim.x * blockDim.x * 8;
  for (; i < total; i += st) {
    long row = i / Sq;  // S % 8 == 0 -> all 8 elems share the row
    float mm = Mrow[row];
    float rl = 1.0f / Lrow[row];
    v8bf v = *(v8bf*)(Sbuf + i);
#pragma unroll
    for (int j = 0; j < 8; ++j)
      v[j] = (__bf16)(__expf((float)v[j] - mm) * rl);
    *(v8bf*)(Sbuf + i) = v;
  }
}

// ---------------------------------------------------------------------------
extern "C" void kernel_launch(void* const* d_in, const int* in_sizes, int n_in,
                              void* d_out, int out_size, void* d_ws, size_t ws_size,
                              hipStream_t stream) {
  (void)in_sizes; (void)n_in; (void)out_size; (void)ws_size;
  const int B = 4, S = 2048, D = 1024;

  const float* x  = (const float*)d_in[0];
  const float* Wq = (const float*)d_in[1];
  const float* bq = (const float*)d_in[2];
  const float* Wk = (const float*)d_in[3];
  const float* bk = (const float*)d_in[4];
  const float* Wv = (const float*)d_in[5];
  const float* bv = (const float*)d_in[6];
  float* out = (float*)d_out;

  char* ws = (char*)d_ws;
  size_t off = 0;
  auto alloc = [&](size_t bytes) -> void* {
    void* p = ws + off;
    off += (bytes + 255) & ~(size_t)255;
    return p;
  };
  const size_t BSD = (size_t)B * S * D;
  __bf16* xbf  = (__bf16*)alloc(BSD * 2);
  __bf16* Wtq  = (__bf16*)alloc((size_t)D * D * 2);
  __bf16* Wtk  = (__bf16*)alloc((size_t)D * D * 2);
  __bf16* Wtv  = (__bf16*)alloc((size_t)D * D * 2);
  __bf16* Qbf  = (__bf16*)alloc(BSD * 2);
  __bf16* Kbf  = (__bf16*)alloc(BSD * 2);
  __bf16* Vbf  = (__bf16*)alloc(BSD * 2);
  __bf16* Vt   = (__bf16*)alloc(BSD * 2);                // [B][D][S]
  __bf16* Sbuf = (__bf16*)alloc((size_t)B * S * S * 2);  // scores -> probs
  float*  Mrow = (float*)alloc((size_t)B * S * 4);
  float*  Lrow = (float*)alloc((size_t)B * S * 4);

  // 1. x -> bf16
  cvt_f32_to_bf16_vec<<<4096, 256, 0, stream>>>(
      (const float4*)x, (v4bf*)xbf, (long)(BSD / 4));

  // 2. W -> Wt bf16 ([out][in])
  dim3 gw(D / 32, D / 32, 1);
  transpose_w_f32_to_bf16<<<gw, 256, 0, stream>>>(Wq, Wtq, D);
  transpose_w_f32_to_bf16<<<gw, 256, 0, stream>>>(Wk, Wtk, D);
  transpose_w_f32_to_bf16<<<gw, 256, 0, stream>>>(Wv, Wtv, D);

  // 3. Q/K/V = xbf @ W + b  (M = B*S = 8192, N = K = D)
  dim3 gqkv((B * S) / 128, D / 64, 1);
  gemm_bf16_wmma<<<gqkv, 256, 0, stream>>>(xbf, D, 0, Wtq, D, 0, bq,
                                           Qbf, nullptr, D, 0, D);
  gemm_bf16_wmma<<<gqkv, 256, 0, stream>>>(xbf, D, 0, Wtk, D, 0, bk,
                                           Kbf, nullptr, D, 0, D);
  gemm_bf16_wmma<<<gqkv, 256, 0, stream>>>(xbf, D, 0, Wtv, D, 0, bv,
                                           Vbf, nullptr, D, 0, D);

  // 4. V -> Vt ([B][D][S], K-contiguous per output column for the P@V GEMM)
  dim3 gvt(D / 32, S / 32, B);
  transpose_bf16_batched<<<gvt, 256, 0, stream>>>(Vbf, Vt, S, D);

  // 5. scores + online softmax stats
  dim3 gsc(S / 128, 1, B);
  attn_scores_bf16<<<gsc, 256, 0, stream>>>(Qbf, Kbf, Sbuf, Mrow, Lrow, S, D);

  // 6. normalize in place
  softmax_normalize<<<8192, 256, 0, stream>>>(Sbuf, Mrow, Lrow, S,
                                              (long)B * S * S);

  // 7. out = P @ V  (per batch: M = S, K = S, N = D, f32 output)
  dim3 gav(S / 128, D / 64, B);
  gemm_bf16_wmma<<<gav, 256, 0, stream>>>(Sbuf, S, (long)S * S,
                                          Vt, S, (long)D * S,
                                          nullptr, nullptr, out, D,
                                          (long)S * D, S);
}
